// QueryAwareRGCN_42013370090000
// MI455X (gfx1250) — matrix operand order repaired and checked by reading, not compiled, not run on hardware
//
#include <hip/hip_runtime.h>
#include <hip/hip_bf16.h>
#include <math.h>

// ---------------- problem constants (match reference) ----------------
#define N_NODES 50000
#define N_EDGES 800000
#define EMB_DIM 128
#define GCN_DIM 128
#define Q_DIM   768
#define Q_PROJ  128
#define N_REL   50
#define N_BASES 8
#define N_LAYERS 3
#define N_QUERIES 8

static_assert(N_NODES % 16 == 0, "M-tiles must not straddle the node count");
static_assert(N_NODES < 65536, "TDM gather uses 16-bit row indices");

#define LDA_T 136            // LDS row stride in bf16 elems (272B) -> conflict-free
#define MAX_STILES (N_EDGES/64 + N_REL)   // 64-edge supertiles upper bound

typedef __attribute__((ext_vector_type(16))) __bf16 v16bf;
typedef __attribute__((ext_vector_type(8)))  __bf16 v8bf;
typedef __attribute__((ext_vector_type(8)))  float  v8f;
typedef __attribute__((ext_vector_type(4)))  unsigned int u32x4;
typedef __attribute__((ext_vector_type(8)))  int i32x8;
typedef __attribute__((ext_vector_type(4)))  int i32x4;

#if __has_builtin(__builtin_amdgcn_tensor_load_to_lds) && \
    __has_builtin(__builtin_amdgcn_s_wait_tensorcnt)
#define HAVE_TDM 1
#pragma message("CDNA5: TDM gather path ENABLED (tensor_load_to_lds)")
#else
#define HAVE_TDM 0
#pragma message("CDNA5: TDM builtins NOT available, using vector-gather fallback")
#endif

// ---------------------------------------------------------------------
// Wave-level 16x16 tile: A(16x128 bf16, LDS, stride LDA_T) @ B from
// preloaded register fragments (4 x v16bf covering K=128 for one N-tile).
// A layout per ISA 7.12.2: lane m=L%16, g=L/16; 16 bf16 per lane are two
// contiguous runs of 8 at K = 32*kt + 8*g and +16.
// ---------------------------------------------------------------------
__device__ __forceinline__ v8f wmma_breg(const __bf16* At, const v16bf* bfrag) {
  const int lane = threadIdx.x & 31;
  const int m = lane & 15;
  const int g = lane >> 4;
  v8f acc = {};
#pragma unroll
  for (int kt = 0; kt < 4; ++kt) {
    const __bf16* ap = At + m * LDA_T + kt * 32 + 8 * g;
    v8bf r1 = *(const v8bf*)ap;
    v8bf r2 = *(const v8bf*)(ap + 16);
    v16bf a;
#pragma unroll
    for (int i = 0; i < 8; ++i) { a[i] = r1[i]; a[i + 8] = r2[i]; }
    acc = __builtin_amdgcn_wmma_f32_16x16x32_bf16(false, a, false, bfrag[kt],
                                                  (short)0, acc, false, false);
  }
  return acc;
}

// B fragments for this wave's N-tile: lane L = column L%16, h = L/16; 16 bf16
// per lane at K = 32*kt + 8*h + (0..7) and +16 (same order as A).
__device__ __forceinline__ void load_bfrag(const __bf16* B, int ncol0, v16bf* bfrag) {
  const int lane = threadIdx.x & 31;
  const int nn = lane & 15, h = lane >> 4;
#pragma unroll
  for (int kt = 0; kt < 4; ++kt)
#pragma unroll
    for (int s = 0; s < 16; ++s)
      bfrag[kt][s] = B[(size_t)(kt * 32 + 8 * h + (s & 7) + ((s >> 3) << 4)) * 128 + ncol0 + nn];
}

// ---------------- small utility kernels ----------------
__global__ void k_zero_f32(float* p, int n) {
  int i = blockIdx.x * blockDim.x + threadIdx.x;
  if (i < n) p[i] = 0.0f;
}
__global__ void k_zero_i32(int* p, int n) {
  int i = blockIdx.x * blockDim.x + threadIdx.x;
  if (i < n) p[i] = 0;
}
__global__ void k_cvt_bf16(const float* in, __bf16* out, int n) {
  int i = blockIdx.x * blockDim.x + threadIdx.x;
  if (i < n) out[i] = (__bf16)in[i];
}

// deg[dst] += 1   (mean-agg denominator; identical across layers)
__global__ void k_deg(const int* edge_index, float* deg) {
  int e = blockIdx.x * blockDim.x + threadIdx.x;
  if (e < N_EDGES) atomicAdd(&deg[edge_index[N_EDGES + e]], 1.0f);
}
__global__ void k_deg_inv(float* deg) {
  int i = blockIdx.x * blockDim.x + threadIdx.x;
  if (i < N_NODES) deg[i] = 1.0f / fmaxf(deg[i], 1.0f);
}

// counting-sort of edges by relation (order within a bin is irrelevant: sum-agg)
__global__ void k_hist(const int* edge_type, int* counts) {
  int e = blockIdx.x * blockDim.x + threadIdx.x;
  if (e < N_EDGES) atomicAdd(&counts[edge_type[e]], 1);
}
__global__ void k_scan(const int* counts, int* rel_off, int* stile_off) {
  if (blockIdx.x == 0 && threadIdx.x == 0) {
    int o = 0, t = 0;
    for (int r = 0; r < N_REL; ++r) {
      rel_off[r] = o; stile_off[r] = t;
      o += counts[r]; t += (counts[r] + 63) >> 6;   // 64-edge supertiles
    }
    rel_off[N_REL] = o; stile_off[N_REL] = t;
  }
}
__global__ void k_scatter(const int* edge_type, const int* rel_off,
                          int* cursors, int* sorted_eid) {
  int e = blockIdx.x * blockDim.x + threadIdx.x;
  if (e >= N_EDGES) return;
  int r = edge_type[e];
  int pos = rel_off[r] + atomicAdd(&cursors[r], 1);
  sorted_eid[pos] = e;
}

// q = query_emb @ query_proj_w + b   (8x768 @ 768x128, tiny -> VALU)
__global__ void k_qproj(const float* qe, const float* w, const float* b, float* q) {
  int t = blockIdx.x * blockDim.x + threadIdx.x;   // 1024 outputs
  if (t >= N_QUERIES * Q_PROJ) return;
  int qi = t >> 7, j = t & 127;
  float s = b[j];
  for (int k = 0; k < Q_DIM; ++k) s += qe[qi * Q_DIM + k] * w[k * Q_PROJ + j];
  q[t] = s;
}

// att_tab[l][qb][r] = sigmoid(q[qb] . rel_attn[l][r])  (3*8*50 = 1200 dots)
__global__ void k_att_tab(const float* q, const float* rel_attn, float* att_tab) {
  int idx = blockIdx.x * blockDim.x + threadIdx.x;
  if (idx >= N_LAYERS * N_QUERIES * N_REL) return;
  int l = idx / (N_QUERIES * N_REL);
  int rem = idx % (N_QUERIES * N_REL);
  int qb = rem / N_REL, r = rem % N_REL;
  const float* qp = q + qb * Q_PROJ;
  const float* ap = rel_attn + ((size_t)l * N_REL + r) * Q_PROJ;
  float s = 0.0f;
  for (int d = 0; d < Q_PROJ; ++d) s += qp[d] * ap[d];
  att_tab[idx] = 1.0f / (1.0f + expf(-s));
}

// W[l][r] = sum_b comp[l][r][b] * bases[l][b]   -> bf16 (150 matrices)
__global__ void k_relw(const float* bases, const float* comp, __bf16* wrel) {
  int blk = blockIdx.x;                 // 0..149
  int l = blk / N_REL, r = blk % N_REL;
  float c[N_BASES];
#pragma unroll
  for (int b = 0; b < N_BASES; ++b) c[b] = comp[((size_t)l * N_REL + r) * N_BASES + b];
  const float* bl = bases + (size_t)l * N_BASES * GCN_DIM * GCN_DIM;
  __bf16* out = wrel + (size_t)blk * GCN_DIM * GCN_DIM;
  for (int e = threadIdx.x; e < GCN_DIM * GCN_DIM; e += blockDim.x) {
    float s = 0.0f;
#pragma unroll
    for (int b = 0; b < N_BASES; ++b) s += c[b] * bl[(size_t)b * GCN_DIM * GCN_DIM + e];
    out[e] = (__bf16)s;
  }
}

// h0 = embed[x] @ input_lin_w + b : gathered-A WMMA GEMM.
// 64 rows per block (4 subtiles), B held in registers across subtiles.
__global__ void k_embed_lin(const int* x, const float* embed, const __bf16* wlin,
                            const float* bias, float* hout) {
  __shared__ __align__(16) __bf16 At[64 * LDA_T];
  const int m0 = blockIdx.x * 64;
  const int t = threadIdx.x;
  const int cseg = (t & 15) * 8;
  const int ntiles = min(4, (N_NODES - m0) >> 4);   // uniform per block
#pragma unroll
  for (int jj = 0; jj < 4; ++jj) {
    int row = jj * 16 + (t >> 4);
    int gm = m0 + row;
    if (gm < N_NODES) {
      const float* sr = embed + (size_t)x[gm] * EMB_DIM + cseg;
#pragma unroll
      for (int j = 0; j < 8; ++j) At[row * LDA_T + cseg + j] = (__bf16)sr[j];
    }
  }
  __syncthreads();
  const int ntile = t >> 5, lane = t & 31, g = lane >> 4;
  v16bf bfrag[4];
  load_bfrag(wlin, ntile * 16, bfrag);
  const int n = ntile * 16 + (lane & 15);
  const float bn = bias[n];
  for (int j = 0; j < ntiles; ++j) {
    v8f acc = wmma_breg(At + j * 16 * LDA_T, bfrag);
#pragma unroll
    for (int i = 0; i < 8; ++i)
      hout[(size_t)(m0 + j * 16 + i + 8 * g) * GCN_DIM + n] = acc[i] + bn;
  }
}

// Relation-grouped edge messages: one 64-edge supertile per block, processed
// as <=4 16-edge subtiles with W_r fragments register-resident across them.
// A-tile = TDM row-gather of hb[src] into LDS (gather-mode D#, hardware-padded
// to the 272B bank-conflict-free stride); m = A @ W_r ; scaled by att ;
// atomic scatter into agg[dst].
__global__ void k_edge_msg(const __bf16* hb, const __bf16* wrel_l,
                           const int* sorted_eid, const int* rel_off,
                           const int* stile_off, const int* edge_index,
                           const int* edge_batch, const float* att_tab_l,
                           float* agg) {
  __shared__ __align__(16) __bf16 At[16 * LDA_T];
  __shared__ float satt[16];
  __shared__ int   sdst[16];
  const int b = blockIdx.x;
  if (b >= stile_off[N_REL]) return;
  int r = 0;
  while (r < N_REL && stile_off[r + 1] <= b) ++r;    // 50-entry table, L2-hot
  const int tin  = b - stile_off[r];
  const int rbeg = rel_off[r], rend = rel_off[r + 1];

  const int t = threadIdx.x;
  const int row = t >> 4, cseg = (t & 15) * 8;
  const int ntile = t >> 5, lane = t & 31, g = lane >> 4;
  v16bf bfrag[4];
  load_bfrag(wrel_l, ntile * 16, bfrag);
  const int n = ntile * 16 + (lane & 15);

  for (int j = 0; j < 4; ++j) {
    const int base = rbeg + tin * 64 + j * 16;
    int cnt = rend - base;
    if (cnt <= 0) break;                 // uniform across block
    if (cnt > 16) cnt = 16;

    if ((t & 15) == 0) {
      if (row < cnt) {
        int e = sorted_eid[base + row];
        sdst[row] = edge_index[N_EDGES + e];
        satt[row] = att_tab_l[edge_batch[e] * N_REL + r];
      } else { sdst[row] = 0; satt[row] = 0.0f; }
    }

#if HAVE_TDM
    if ((t >> 5) == 0) {  // wave 0 only: TDM ignores EXEC -> gate wave-uniformly
      // 16 uniform row indices (rows >= cnt reuse row 0; nulled via att=0)
      unsigned idx[16];
#pragma unroll
      for (int i = 0; i < 16; ++i) {
        int e = sorted_eid[base + ((i < cnt) ? i : 0)];
        idx[i] = (unsigned)edge_index[e];
      }
      const unsigned lds_off = (unsigned)(uintptr_t)(void*)At;
      const unsigned long long ga = (unsigned long long)(uintptr_t)hb;
      u32x4 g0;
      g0[0] = 0x80000001u;                           // count=1, gather_mode=1, 16b idx
      g0[1] = lds_off;                               // LDS destination
      g0[2] = (unsigned)ga;                          // global_addr[31:0]
      g0[3] = ((unsigned)(ga >> 32) & 0x01FFFFFFu) | 0x80000000u; // addr|type=2
      i32x8 g1;
      g1[0] = (int)((1u << 16)       // data_size = 2B
                  | (1u << 20)       // pad_enable
                  | (5u << 22)       // pad_interval: 64 DWORDs (256B row)
                  | (3u << 25));     // pad_amount: 4 DWORDs (16B) -> 272B stride
      g1[1] = (int)((unsigned)GCN_DIM << 16);           // tensor_dim0 = 128
      g1[2] = (int)(((unsigned)N_NODES & 0xFFFFu) << 16); // tensor_dim1 = N_NODES
      g1[3] = (int)((unsigned)GCN_DIM << 16);           // tile_dim0 = 128
      g1[4] = 16;                                       // tile_dim1 = 16 indices
      g1[5] = GCN_DIM;                                  // tensor_dim0_stride
      g1[6] = 0; g1[7] = 0;
      i32x4 g2, g3;
#pragma unroll
      for (int i = 0; i < 4; ++i) {
        g2[i] = (int)(idx[2 * i]     | (idx[2 * i + 1] << 16));
        g3[i] = (int)(idx[8 + 2 * i] | (idx[9 + 2 * i] << 16));
      }
#if defined(__clang_major__) && __clang_major__ >= 23
      i32x8 gx = {};
      __builtin_amdgcn_tensor_load_to_lds(g0, g1, g2, g3, gx, 0);
#else
      __builtin_amdgcn_tensor_load_to_lds(g0, g1, g2, g3, 0);
#endif
      __builtin_amdgcn_s_wait_tensorcnt(0);
    }
#else
    if (row < cnt) {
      int e = sorted_eid[base + row];
      int s = edge_index[e];
      *(v8bf*)&At[row * LDA_T + cseg] =
          *(const v8bf*)(hb + (size_t)s * GCN_DIM + cseg);
    } else {
      v8bf z = {};
      *(v8bf*)&At[row * LDA_T + cseg] = z;
    }
#endif
    __syncthreads();

    v8f acc = wmma_breg(At, bfrag);
#pragma unroll
    for (int i = 0; i < 8; ++i) {
      int rr = i + 8 * g;
      atomicAdd(&agg[(size_t)sdst[rr] * GCN_DIM + n], acc[i] * satt[rr]);
    }
    __syncthreads();   // protect At/satt/sdst before next subtile overwrites
  }
}

// h_next = agg * dinv + h @ root_w + bias  (+relu)
// 64 rows per block, register-resident B across 4 subtiles.
__global__ void k_combine(const __bf16* hb, const __bf16* wroot_l, const float* agg,
                          const float* dinv, const float* bias_l, float* hout,
                          int do_relu) {
  __shared__ __align__(16) __bf16 At[64 * LDA_T];
  const int m0 = blockIdx.x * 64;
  const int t = threadIdx.x;
  const int cseg = (t & 15) * 8;
  const int ntiles = min(4, (N_NODES - m0) >> 4);   // uniform per block
#pragma unroll
  for (int jj = 0; jj < 4; ++jj) {
    int row = jj * 16 + (t >> 4);
    int gm = m0 + row;
    if (gm < N_NODES)
      *(v8bf*)&At[row * LDA_T + cseg] =
          *(const v8bf*)(hb + (size_t)gm * GCN_DIM + cseg);
  }
  __syncthreads();
  const int ntile = t >> 5, lane = t & 31, g = lane >> 4;
  v16bf bfrag[4];
  load_bfrag(wroot_l, ntile * 16, bfrag);
  const int n = ntile * 16 + (lane & 15);
  const float bn = bias_l[n];
  for (int j = 0; j < ntiles; ++j) {
    v8f acc = wmma_breg(At + j * 16 * LDA_T, bfrag);
#pragma unroll
    for (int i = 0; i < 8; ++i) {
      int mr = m0 + j * 16 + i + 8 * g;
      float v = acc[i] + bn + agg[(size_t)mr * GCN_DIM + n] * dinv[mr];
      if (do_relu) v = fmaxf(v, 0.0f);
      hout[(size_t)mr * GCN_DIM + n] = v;
    }
  }
}

// ---------------------------------------------------------------------
extern "C" void kernel_launch(void* const* d_in, const int* in_sizes, int n_in,
                              void* d_out, int out_size, void* d_ws, size_t ws_size,
                              hipStream_t stream) {
  const int*   x            = (const int*)  d_in[0];
  const int*   edge_index   = (const int*)  d_in[1];   // [2, E] flat
  const int*   edge_type    = (const int*)  d_in[2];
  const float* query_emb    = (const float*)d_in[3];
  /* x_batch d_in[4] unused */
  const int*   edge_batch   = (const int*)  d_in[5];
  const float* embed_weight = (const float*)d_in[6];
  const float* input_lin_w  = (const float*)d_in[7];
  const float* input_lin_b  = (const float*)d_in[8];
  const float* query_proj_w = (const float*)d_in[9];
  const float* query_proj_b = (const float*)d_in[10];
  const float* bases        = (const float*)d_in[11];
  const float* comp         = (const float*)d_in[12];
  const float* root_w       = (const float*)d_in[13];
  const float* layer_bias   = (const float*)d_in[14];
  const float* rel_attn     = (const float*)d_in[15];
  float* out = (float*)d_out;

  // ---- workspace carve-up (bump allocator, 256B aligned; ~97 MB total) ----
  char* p = (char*)d_ws;
  auto alloc = [&](size_t bytes) -> void* {
    void* r = (void*)p;
    p += (bytes + 255) & ~(size_t)255;
    return r;
  };
  float*  hA        = (float*) alloc((size_t)N_NODES * GCN_DIM * 4);
  float*  hB        = (float*) alloc((size_t)N_NODES * GCN_DIM * 4);
  __bf16* hb        = (__bf16*)alloc((size_t)N_NODES * GCN_DIM * 2);
  float*  agg       = (float*) alloc((size_t)N_NODES * GCN_DIM * 4);
  float*  dinv      = (float*) alloc((size_t)N_NODES * 4);
  float*  q         = (float*) alloc((size_t)N_QUERIES * Q_PROJ * 4);
  float*  att_tab   = (float*) alloc((size_t)N_LAYERS * N_QUERIES * N_REL * 4);
  __bf16* wrel      = (__bf16*)alloc((size_t)N_LAYERS * N_REL * GCN_DIM * GCN_DIM * 2);
  __bf16* wlin      = (__bf16*)alloc((size_t)GCN_DIM * GCN_DIM * 2);
  __bf16* wroot     = (__bf16*)alloc((size_t)N_LAYERS * GCN_DIM * GCN_DIM * 2);
  int*    sorted    = (int*)   alloc((size_t)N_EDGES * 4);
  int*    counts    = (int*)   alloc(64 * 4);
  int*    cursors   = (int*)   alloc(64 * 4);
  int*    rel_off   = (int*)   alloc(64 * 4);
  int*    stile_off = (int*)   alloc(64 * 4);

  const int T = 256;
  const int NF = N_NODES * GCN_DIM;
  dim3 blk(T);
  const int mblocks = (N_NODES + 63) / 64;

  // ---- one-time preprocessing (re-run every call; deterministic) ----
  k_cvt_bf16<<<(GCN_DIM*GCN_DIM + T-1)/T, blk, 0, stream>>>(input_lin_w, wlin, GCN_DIM*GCN_DIM);
  k_cvt_bf16<<<(N_LAYERS*GCN_DIM*GCN_DIM + T-1)/T, blk, 0, stream>>>(root_w, wroot, N_LAYERS*GCN_DIM*GCN_DIM);
  k_relw<<<N_LAYERS * N_REL, blk, 0, stream>>>(bases, comp, wrel);
  k_qproj<<<(N_QUERIES*Q_PROJ + T-1)/T, blk, 0, stream>>>(query_emb, query_proj_w, query_proj_b, q);
  k_att_tab<<<(N_LAYERS*N_QUERIES*N_REL + T-1)/T, blk, 0, stream>>>(q, rel_attn, att_tab);

  k_zero_f32<<<(N_NODES + T-1)/T, blk, 0, stream>>>(dinv, N_NODES);
  k_deg<<<(N_EDGES + T-1)/T, blk, 0, stream>>>(edge_index, dinv);
  k_deg_inv<<<(N_NODES + T-1)/T, blk, 0, stream>>>(dinv);

  k_zero_i32<<<1, blk, 0, stream>>>(counts, 64);
  k_zero_i32<<<1, blk, 0, stream>>>(cursors, 64);
  k_hist<<<(N_EDGES + T-1)/T, blk, 0, stream>>>(edge_type, counts);
  k_scan<<<1, 32, 0, stream>>>(counts, rel_off, stile_off);
  k_scatter<<<(N_EDGES + T-1)/T, blk, 0, stream>>>(edge_type, rel_off, cursors, sorted);

  // ---- h0 = embed[x] @ input_lin_w + b ----
  k_embed_lin<<<mblocks, blk, 0, stream>>>(x, embed_weight, wlin, input_lin_b, hA);

  // ---- 3 conv layers ----
  float* hcur = hA;
  float* hnxt = hB;
  for (int l = 0; l < N_LAYERS; ++l) {
    k_cvt_bf16<<<(NF + T-1)/T, blk, 0, stream>>>(hcur, hb, NF);
    k_zero_f32<<<(NF + T-1)/T, blk, 0, stream>>>(agg, NF);
    k_edge_msg<<<MAX_STILES, blk, 0, stream>>>(
        hb, wrel + (size_t)l * N_REL * GCN_DIM * GCN_DIM,
        sorted, rel_off, stile_off, edge_index, edge_batch,
        att_tab + (size_t)l * N_QUERIES * N_REL, agg);
    float* dst = (l == N_LAYERS - 1) ? out : hnxt;
    k_combine<<<mblocks, blk, 0, stream>>>(
        hb, wroot + (size_t)l * GCN_DIM * GCN_DIM, agg, dinv,
        layer_bias + (size_t)l * GCN_DIM, dst, (l != N_LAYERS - 1) ? 1 : 0);
    float* tmp = hcur; hcur = hnxt; hnxt = tmp;
    if (l == 0) hnxt = hA;   // after first swap keep ping-ponging A/B
  }
}
